// SupConNTXentLoss_46016279609997
// MI455X (gfx1250) — matrix operand order repaired and checked by reading, not compiled, not run on hardware
//
#include <hip/hip_runtime.h>
#include <hip/hip_bf16.h>
#include <math.h>
#include <stdint.h>

// ---------------------------------------------------------------------------
// SupCon NT-Xent loss, N=4096, E=256, T=0.1, fused CDNA5 WMMA implementation.
//   sim = zn @ zn^T computed in 32x16 tiles per wave with
//   v_wmma_f32_16x16x32_bf16 using a bf16 hi/lo split (3 GEMM passes ~ fp32
//   accuracy); sim is never materialized. B tiles are staged into LDS by the
//   Tensor Data Mover (tensor_load_to_lds, double buffered, TENSORcnt waits)
//   when the builtin is available, else by cooperative ds_store_b128.
// ---------------------------------------------------------------------------

#define NHALF   4096          // N
#define NROWS   8192          // 2N
#define EDIM    256           // embedding dim
#define LDSPITCH 264          // bf16 elements per staged row (256 + 8 pad)
#define SLICE_TILES 32        // 16-col tiles per column slice (512 cols)
#define LOG2E_OVER_T 14.4269504088896340736f   // (1/T) * log2(e), T = 0.1

typedef __attribute__((ext_vector_type(16))) __bf16 v16bf;
typedef __attribute__((ext_vector_type(8)))  float  v8f;

union BF16x16 {           // one 16x16x32 bf16 WMMA fragment per lane (32 B)
    v16bf v;
    uint4 q[2];
};

// ---------------------------------------------------------------------------
// Tensor Data Mover staging (CDNA5): one 2D descriptor moves a 16-row x 512 B
// tile into LDS with a 16 B pad after every 512 B row (pitch 528 B = LDSPITCH).
// ---------------------------------------------------------------------------
#if defined(__gfx1250__) && __has_builtin(__builtin_amdgcn_tensor_load_to_lds)
#define USE_TDM 1
typedef __attribute__((ext_vector_type(4))) unsigned int u32x4;
typedef __attribute__((ext_vector_type(8))) int i32x8;
typedef __attribute__((ext_vector_type(4))) int i32x4;

__device__ __forceinline__ void tdm_load_tile16(const __bf16* gsrc, unsigned ldsDst)
{
    const unsigned long long ga = (unsigned long long)(uintptr_t)gsrc;
    u32x4 g0;
    g0[0] = 1u;                                   // count=1, user descriptor
    g0[1] = ldsDst;                               // lds_addr (bytes)
    g0[2] = (unsigned)ga;                         // global_addr[31:0]
    g0[3] = (unsigned)((ga >> 32) & 0x01FFFFFFull) | (2u << 30);  // addr[56:32] | type=2
    i32x8 g1;
    g1[0] = (int)((1u << 16)      // data_size = 2 bytes
                | (1u << 20)      // pad_enable
                | (6u << 22)      // pad_interval: 128 dwords (512 B row)
                | (3u << 25));    // pad_amount: 4 dwords (16 B)
    g1[1] = (int)(256u << 16);                    // tensor_dim0 = 256 elems
    g1[2] = (int)((8192u & 0xFFFFu) << 16);       // tensor_dim1 = 8192 rows
    g1[3] = (int)(256u << 16);                    // tile_dim0 = 256 elems
    g1[4] = 16;                                   // tile_dim1 = 16 rows, tile_dim2 = 0
    g1[5] = 256;                                  // tensor_dim0_stride = 256 elems
    g1[6] = 0;
    g1[7] = 0;
    const i32x4 z4 = {0, 0, 0, 0};
#if __clang_major__ >= 23
    const i32x8 z8 = {0, 0, 0, 0, 0, 0, 0, 0};
    __builtin_amdgcn_tensor_load_to_lds(g0, g1, z4, z4, z8, 0);
#else
    __builtin_amdgcn_tensor_load_to_lds(g0, g1, z4, z4, 0);
#endif
}
#endif

// ---------------------------------------------------------------------------
// Kernel 1: row-normalize z = concat(tab, ts) and split into bf16 hi/lo planes
// ---------------------------------------------------------------------------
__global__ __launch_bounds__(256) void supcon_normalize(
    const float* __restrict__ tab, const float* __restrict__ ts,
    __bf16* __restrict__ hiP, __bf16* __restrict__ loP)
{
    __shared__ float red[256];
    const int row = blockIdx.x;                  // 0..8191
    const int t   = threadIdx.x;                 // 0..255 == column
    const float* src = (row < NHALF) ? (tab + (long)row * EDIM)
                                     : (ts  + (long)(row - NHALF) * EDIM);
    const float x = src[t];
    red[t] = x * x;
    __syncthreads();
    for (int s = 128; s > 0; s >>= 1) {
        if (t < s) red[t] += red[t + s];
        __syncthreads();
    }
    float nrm = fmaxf(sqrtf(red[0]), 1e-8f);
    const float xn = x / nrm;
    const __bf16 h = (__bf16)xn;
    const __bf16 l = (__bf16)(xn - (float)h);
    hiP[(long)row * EDIM + t] = h;
    loP[(long)row * EDIM + t] = l;
}

// ---------------------------------------------------------------------------
// Kernel 2: zero the accumulators and the output scalar
// ---------------------------------------------------------------------------
__global__ void supcon_init(float* __restrict__ denom, float* __restrict__ psum,
                            float* __restrict__ out)
{
    const int i = blockIdx.x * blockDim.x + threadIdx.x;
    if (i < NROWS) { denom[i] = 0.0f; psum[i] = 0.0f; }
    if (i == 0)    out[0] = 0.0f;
}

// ---------------------------------------------------------------------------
// Kernel 3: fused sim-tile GEMM (bf16x3 WMMA) + exp/mask row reductions.
// Block = 256 threads = 8 waves, covers 256 rows x 512-column slice.
// Each wave computes a 32x16 output (2 row tiles, 2 independent WMMA chains),
// sharing each LDS-staged 16-column B tile across all 8 waves.
// ---------------------------------------------------------------------------
__global__ __launch_bounds__(256) void supcon_main(
    const __bf16* __restrict__ hiP, const __bf16* __restrict__ loP,
    const int* __restrict__ labels,
    float* __restrict__ denom, float* __restrict__ psum)
{
    __shared__ __align__(16) __bf16 ldsB[2][2][16 * LDSPITCH]; // [buf][hi/lo]

    const int tid  = threadIdx.x;
    const int lane = tid & 31;                   // wave32
    const int wave = tid >> 5;                   // 0..7
    const int g    = lane >> 4;                  // half-wave: 0/1
    const int ln   = lane & 15;

    const int r0     = blockIdx.x * 256 + wave * 32;      // wave's 32 rows
    const int j0base = blockIdx.y * (SLICE_TILES * 16);   // 512-col slice

    // ---- preload A fragments (2 row tiles x 256 k, hi+lo) into registers --
    // A layout (16-bit, 16x32): lane L holds M=L%16; VGPR v: K=(v<4?2v:2v+8)+8g
    BF16x16 aHi[2][8], aLo[2][8];
#pragma unroll
    for (int rt = 0; rt < 2; ++rt) {
        const long rowOff = (long)(r0 + rt * 16 + ln) * EDIM;
#pragma unroll
        for (int ks = 0; ks < 8; ++ks) {
            const int kb = ks * 32;
            const int o0 = kb + 8 * g;
            const int o1 = kb + 16 + 8 * g;
            aHi[rt][ks].q[0] = *reinterpret_cast<const uint4*>(hiP + rowOff + o0);
            aHi[rt][ks].q[1] = *reinterpret_cast<const uint4*>(hiP + rowOff + o1);
            aLo[rt][ks].q[0] = *reinterpret_cast<const uint4*>(loP + rowOff + o0);
            aLo[rt][ks].q[1] = *reinterpret_cast<const uint4*>(loP + rowOff + o1);
        }
    }

    int rowLab[2][8];
    float dAcc[2][8], pAcc[2][8];
#pragma unroll
    for (int rt = 0; rt < 2; ++rt)
#pragma unroll
        for (int v = 0; v < 8; ++v) {
            rowLab[rt][v] = labels[(r0 + rt * 16 + v + 8 * g) & (NHALF - 1)];
            dAcc[rt][v] = 0.0f;
            pAcc[rt][v] = 0.0f;
        }

#if defined(USE_TDM)
    if (wave == 0) {   // prologue: DMA tile 0 (hi+lo planes) into buffer 0
        tdm_load_tile16(hiP + (size_t)j0base * EDIM, (unsigned)(uintptr_t)&ldsB[0][0][0]);
        tdm_load_tile16(loP + (size_t)j0base * EDIM, (unsigned)(uintptr_t)&ldsB[0][1][0]);
    }
#endif

    for (int t = 0; t < SLICE_TILES; ++t) {
        const int buf = t & 1;
        const int j0  = j0base + t * 16;

#if defined(USE_TDM)
        if (wave == 0) __builtin_amdgcn_s_wait_tensorcnt(0);  // tile t landed
        __syncthreads();            // LDS visible to all; prior reads of buf^1 done
        if (wave == 0 && (t + 1) < SLICE_TILES) {             // overlap DMA t+1
            const size_t nOff = (size_t)(j0 + 16) * EDIM;
            tdm_load_tile16(hiP + nOff, (unsigned)(uintptr_t)&ldsB[buf ^ 1][0][0]);
            tdm_load_tile16(loP + nOff, (unsigned)(uintptr_t)&ldsB[buf ^ 1][1][0]);
        }
#else
        {   // cooperative stage of B tile (rows j0..j0+15, both planes)
            const int rr  = tid >> 4;            // 0..15 : staged row
            const int seg = tid & 15;            // 16 bf16 (32 B) per thread
            const long gOff = (long)(j0 + rr) * EDIM + seg * 16;
            const int  lOff = rr * LDSPITCH + seg * 16;
            *reinterpret_cast<uint4*>(&ldsB[buf][0][lOff])     = *reinterpret_cast<const uint4*>(hiP + gOff);
            *reinterpret_cast<uint4*>(&ldsB[buf][0][lOff + 8]) = *reinterpret_cast<const uint4*>(hiP + gOff + 8);
            *reinterpret_cast<uint4*>(&ldsB[buf][1][lOff])     = *reinterpret_cast<const uint4*>(loP + gOff);
            *reinterpret_cast<uint4*>(&ldsB[buf][1][lOff + 8]) = *reinterpret_cast<const uint4*>(loP + gOff + 8);
        }
        __syncthreads();
#endif

        // ---- two 16x16 sim tiles: 8 k-steps x 6 WMMAs, 2 indep chains -----
        // B layout (16-bit, 32x16): lane L holds N=L%16; K=(v<4?2v:2v+8)+8g
        v8f acc0 = {};
        v8f acc1 = {};
#pragma unroll
        for (int ks = 0; ks < 8; ++ks) {
            const int bo = ln * LDSPITCH + ks * 32 + 8 * g;
            BF16x16 bHi, bLo;
            bHi.q[0] = *reinterpret_cast<const uint4*>(&ldsB[buf][0][bo]);
            bHi.q[1] = *reinterpret_cast<const uint4*>(&ldsB[buf][0][bo + 16]);
            bLo.q[0] = *reinterpret_cast<const uint4*>(&ldsB[buf][1][bo]);
            bLo.q[1] = *reinterpret_cast<const uint4*>(&ldsB[buf][1][bo + 16]);
            acc0 = __builtin_amdgcn_wmma_f32_16x16x32_bf16(false, aHi[0][ks].v, false, bHi.v,
                                                           (short)0, acc0, false, false);
            acc1 = __builtin_amdgcn_wmma_f32_16x16x32_bf16(false, aHi[1][ks].v, false, bHi.v,
                                                           (short)0, acc1, false, false);
            acc0 = __builtin_amdgcn_wmma_f32_16x16x32_bf16(false, aHi[0][ks].v, false, bLo.v,
                                                           (short)0, acc0, false, false);
            acc1 = __builtin_amdgcn_wmma_f32_16x16x32_bf16(false, aHi[1][ks].v, false, bLo.v,
                                                           (short)0, acc1, false, false);
            acc0 = __builtin_amdgcn_wmma_f32_16x16x32_bf16(false, aLo[0][ks].v, false, bHi.v,
                                                           (short)0, acc0, false, false);
            acc1 = __builtin_amdgcn_wmma_f32_16x16x32_bf16(false, aLo[1][ks].v, false, bHi.v,
                                                           (short)0, acc1, false, false);
        }
#if !defined(USE_TDM)
        __syncthreads();   // fallback: B reads done before next tile's ds_store
#endif

        // ---- epilogue: C layout lane L, slot v -> (M = v+8g, N = L%16) ----
        const int  jg      = j0 + ln;
        const int  colLab  = labels[jg & (NHALF - 1)];
        const bool colLeft = (jg < NHALF);
#pragma unroll
        for (int rt = 0; rt < 2; ++rt) {
#pragma unroll
            for (int v = 0; v < 8; ++v) {
                const int   i = r0 + rt * 16 + v + 8 * g;
                const float s = rt ? acc1[v] : acc0[v];
                const float e = (i == jg) ? 0.0f : exp2f(s * LOG2E_OVER_T);
                dAcc[rt][v] += e;                               // diag excluded
                const bool cross = (i < NHALF) != colLeft;      // cross-view
                if (cross && (rowLab[rt][v] == colLab)) pAcc[rt][v] += s;
            }
        }
    }

    // ---- reduce each row over its 16 lanes, then global accumulate --------
#pragma unroll
    for (int rt = 0; rt < 2; ++rt) {
#pragma unroll
        for (int v = 0; v < 8; ++v) {
            float d = dAcc[rt][v];
            float p = pAcc[rt][v];
            for (int off = 8; off >= 1; off >>= 1) {
                d += __shfl_xor(d, off, 16);
                p += __shfl_xor(p, off, 16);
            }
            if (ln == 0) {
                const int i = r0 + rt * 16 + v + 8 * g;
                atomicAdd(&denom[i], d);
                atomicAdd(&psum[i],  p);
            }
        }
    }
}

// ---------------------------------------------------------------------------
// Kernel 4: loss = -(1/2N) * sum_i (10*P_i - N*log(denom_i)) / M_i,
// with M_i = 2 * count(labels == label_i). One block per row.
// ---------------------------------------------------------------------------
__global__ __launch_bounds__(256) void supcon_finalize(
    const int* __restrict__ labels, const float* __restrict__ denom,
    const float* __restrict__ psum, float* __restrict__ out)
{
    __shared__ int red[256];
    const int i  = blockIdx.x;                   // 0..8191
    const int li = labels[i & (NHALF - 1)];
    int c = 0;
    for (int j = threadIdx.x; j < NHALF; j += 256) c += (labels[j] == li) ? 1 : 0;
    red[threadIdx.x] = c;
    __syncthreads();
    for (int s = 128; s > 0; s >>= 1) {
        if (threadIdx.x < s) red[threadIdx.x] += red[threadIdx.x + s];
        __syncthreads();
    }
    if (threadIdx.x == 0) {
        const float M   = 2.0f * (float)red[0];
        const float val = (10.0f * psum[i] - (float)NHALF * logf(denom[i])) / M;
        atomicAdd(out, -val / (float)NROWS);
    }
}

// ---------------------------------------------------------------------------
extern "C" void kernel_launch(void* const* d_in, const int* in_sizes, int n_in,
                              void* d_out, int out_size, void* d_ws, size_t ws_size,
                              hipStream_t stream)
{
    (void)in_sizes; (void)n_in; (void)out_size; (void)ws_size;
    const float* tab    = (const float*)d_in[0];   // (4096, 256) f32
    const float* ts     = (const float*)d_in[1];   // (4096, 256) f32
    const int*   labels = (const int*)d_in[2];     // (4096,) i32
    float*       out    = (float*)d_out;           // scalar f32

    // workspace: hi plane (4 MB) | lo plane (4 MB) | denom (32 KB) | P (32 KB)
    char*   ws    = (char*)d_ws;
    __bf16* hiP   = (__bf16*)ws;
    __bf16* loP   = (__bf16*)(ws + (size_t)NROWS * EDIM * sizeof(__bf16));
    float*  denom = (float*)(ws + (size_t)NROWS * EDIM * sizeof(__bf16) * 2);
    float*  psum  = denom + NROWS;

    supcon_normalize<<<NROWS, 256, 0, stream>>>(tab, ts, hiP, loP);
    supcon_init<<<(NROWS + 255) / 256, 256, 0, stream>>>(denom, psum, out);
    dim3 grid(32, 16);  // 32 row-groups of 256 x 16 column slices of 512
    supcon_main<<<grid, 256, 0, stream>>>(hiP, loP, labels, denom, psum);
    supcon_finalize<<<NROWS, 256, 0, stream>>>(labels, denom, psum, out);
}
